// Word2VecPy_48438641164885
// MI455X (gfx1250) — matrix operand order repaired and checked by reading, not compiled, not run on hardware
//
#include <hip/hip_runtime.h>
#include <hip/hip_bf16.h>
#include <math.h>

// ---- CDNA5 (gfx1250) WMMA vector types ----
typedef __attribute__((ext_vector_type(16))) _Float16 v16h;
typedef __attribute__((ext_vector_type(8)))  _Float16 half8;
typedef __attribute__((ext_vector_type(4)))  _Float16 half4;
typedef __attribute__((ext_vector_type(8)))  float    v8f;

#define D_DIM   300   // embedding dim (divisible by 4 -> full float4 gathers)
#define C_CTX   10    // context window
#define NROWS   60    // ctx (10) + neg (50)
#define RPAD    64    // rows padded to 4 WMMA M-tiles
#define KPAD    320   // K padded to 10 steps of 32
#define AS      328   // LDS row stride in halves (656 B: 16B aligned, bank-conflict-free)
#define G4      (AS / 4)   // 82 float4-groups per padded row

__device__ __forceinline__ float logsigf(float x) {
    // numerically stable log(sigmoid(x)) = min(x,0) - log1p(exp(-|x|))
    return fminf(x, 0.0f) - log1pf(expf(-fabsf(x)));
}

__global__ __launch_bounds__(128)
void w2v_nce_kernel(const int* __restrict__ target,
                    const int* __restrict__ context,
                    const int* __restrict__ negs,
                    const float* __restrict__ emb_in,
                    const float* __restrict__ emb_out,
                    float* __restrict__ ws)
{
    __shared__ __attribute__((aligned(16))) _Float16 Alds[RPAD * AS]; // 60 gathered rows (f16), zero padded
    __shared__ __attribute__((aligned(16))) _Float16 Tlds[AS];        // target row (f16), zero padded
    __shared__ int   idxlds[RPAD];
    __shared__ float partials[8];

    const int b   = blockIdx.x;
    const int tid = threadIdx.x;

    // ---- gather indices for the 60 rows ----
    if (tid < RPAD) {
        int idx = 0;
        if (tid < C_CTX)      idx = context[b * C_CTX + tid];
        else if (tid < NROWS) idx = negs[b * (NROWS - C_CTX) + (tid - C_CTX)];
        idxlds[tid] = idx;
    }
    __syncthreads();

    // ---- stage target row into LDS (float4 global loads, f32 -> f16) ----
    if (tid < G4) {
        const int c4 = tid * 4;
        float4 v = make_float4(0.f, 0.f, 0.f, 0.f);
        if (c4 < D_DIM)  // D_DIM % 4 == 0, so in-range groups are always full
            v = *(const float4*)(emb_in + (size_t)target[b] * D_DIM + c4);
        half4 h = { (_Float16)v.x, (_Float16)v.y, (_Float16)v.z, (_Float16)v.w };
        *(half4*)(&Tlds[c4]) = h;
    }

    // ---- stage 60 gathered emb_out rows into LDS: global_load_b128 + ds_store_b64 ----
    for (int i = tid; i < RPAD * G4; i += 128) {
        const int row = i / G4;
        const int c4  = (i - row * G4) * 4;
        float4 v = make_float4(0.f, 0.f, 0.f, 0.f);
        if (row < NROWS && c4 < D_DIM)
            v = *(const float4*)(emb_out + (size_t)idxlds[row] * D_DIM + c4);
        half4 h = { (_Float16)v.x, (_Float16)v.y, (_Float16)v.z, (_Float16)v.w };
        *(half4*)(&Alds[row * AS + c4]) = h;   // 8B-aligned: 656B row stride, c4*2 % 8 == 0
    }
    __syncthreads();

    // ---- WMMA: wave w computes scores for rows 16w..16w+15 ----
    const int wave = tid >> 5;        // wave32: 4 waves per block
    const int lane = tid & 31;
    const int sel  = lane >> 4;       // lane half selects K chunk
    const int mrow = lane & 15;       // A: lane%16 = matrix row within tile
    const _Float16* arow = &Alds[(wave * 16 + mrow) * AS];

    v8f acc = {};
    #pragma unroll
    for (int k0 = 0; k0 < KPAD; k0 += 32) {
        // A fragment (16-bit A 16x32 layout): lane<16 -> K 0..7 & 16..23; lane>=16 -> K 8..15 & 24..31
        half8 a0 = *(const half8*)(arow + k0 + sel * 8);
        half8 a1 = *(const half8*)(arow + k0 + sel * 8 + 16);
        // B fragment (32x16, tgt replicated over all N): lanes 0-15 -> K 0..7 & 16..23; lanes 16-31 -> K 8..15 & 24..31
        half8 b0 = *(const half8*)(&Tlds[k0 + sel * 8]);
        half8 b1 = *(const half8*)(&Tlds[k0 + sel * 8 + 16]);
        v16h Af, Bf;
        #pragma unroll
        for (int j = 0; j < 8; ++j) {
            Af[j] = a0[j]; Af[j + 8] = a1[j];
            Bf[j] = b0[j]; Bf[j + 8] = b1[j];
        }
        acc = __builtin_amdgcn_wmma_f32_16x16x32_f16(
            /*neg_a=*/false, Af, /*neg_b=*/false, Bf,
            /*c_mod=*/(short)0, acc, /*reuse_a=*/false, /*reuse_b=*/false);
    }

    // D layout: all N columns identical; this lane-half holds rows 16w + 8*sel + r in acc[r]
    float part = 0.0f;
    #pragma unroll
    for (int r = 0; r < 8; ++r) {
        int g = wave * 16 + sel * 8 + r;
        float s = acc[r];
        if (g < C_CTX)        part += 0.1f * logsigf(s);    // positive: logsig(dot)/C
        else if (g < NROWS)   part += 0.1f * logsigf(-s);   // negative: logsig(-dot)/C
        // pad rows (g >= 60) contribute nothing
    }

    if ((lane & 15) == 0) partials[wave * 2 + sel] = part;
    __syncthreads();

    if (tid == 0) {
        float s = 0.0f;
        #pragma unroll
        for (int i = 0; i < 8; ++i) s += partials[i];  // fixed order -> deterministic
        ws[b] = s;
    }
}

__global__ __launch_bounds__(256)
void w2v_reduce_kernel(const float* __restrict__ ws, int n, float* __restrict__ out)
{
    __shared__ float sm[256];
    int t = threadIdx.x;
    float s = 0.0f;
    for (int i = t; i < n; i += 256) s += ws[i];   // fixed per-thread order
    sm[t] = s;
    __syncthreads();
    for (int off = 128; off > 0; off >>= 1) {       // fixed tree -> deterministic
        if (t < off) sm[t] += sm[t + off];
        __syncthreads();
    }
    if (t == 0) out[0] = -(sm[0] / (float)n);
}

extern "C" void kernel_launch(void* const* d_in, const int* in_sizes, int n_in,
                              void* d_out, int out_size, void* d_ws, size_t ws_size,
                              hipStream_t stream) {
    const int*   target  = (const int*)d_in[0];
    const int*   context = (const int*)d_in[1];
    const int*   negs    = (const int*)d_in[2];
    const float* emb_in  = (const float*)d_in[3];
    const float* emb_out = (const float*)d_in[4];
    float* out = (float*)d_out;
    float* ws  = (float*)d_ws;          // B floats of per-batch partial loss
    const int B = in_sizes[0];

    w2v_nce_kernel<<<B, 128, 0, stream>>>(target, context, negs, emb_in, emb_out, ws);
    w2v_reduce_kernel<<<1, 256, 0, stream>>>(ws, B, out);
}
